// Head_81913616269665
// MI455X (gfx1250) — matrix operand order repaired and checked by reading, not compiled, not run on hardware
//
#include <hip/hip_runtime.h>

// ---------------------------------------------------------------------------
// Head kernel for MI455X (gfx1250): out = scale * Q @ (K^T @ V), fused QKV.
// B=4, T=4096, E=1024, H=64.  bf16 WMMA (16x16x32) with fp32 accumulation.
// Weight slabs staged through LDS with gfx1250 async global->LDS copies.
// ---------------------------------------------------------------------------

typedef __bf16 bf16_t;
typedef bf16_t v16bf __attribute__((ext_vector_type(16)));
typedef float  v8f   __attribute__((ext_vector_type(8)));
typedef int    v4i   __attribute__((ext_vector_type(4)));
typedef __attribute__((address_space(1))) v4i gv4i;   // global int4
typedef __attribute__((address_space(3))) v4i lv4i;   // LDS int4

static constexpr int B_ = 4;
static constexpr int T_ = 4096;
static constexpr int E_ = 1024;
static constexpr int H_ = 64;
static constexpr int M_ = B_ * T_;      // 16384 rows total

// workspace layout (bytes)
static constexpr size_t WT_OFF = 0;                                   // bf16 [192][1024]
static constexpr size_t WT_BYTES = (size_t)3 * H_ * E_ * 2;           // 393216
static constexpr size_t Q_OFF = WT_OFF + WT_BYTES;                    // bf16 [M][64]
static constexpr size_t QKV_BYTES = (size_t)M_ * H_ * 2;              // 2 MB each
static constexpr size_t K_OFF = Q_OFF + QKV_BYTES;
static constexpr size_t V_OFF = K_OFF + QKV_BYTES;
static constexpr size_t S_OFF = V_OFF + QKV_BYTES;                    // f32 [B][64][64]

#if defined(__AMDGCN__) && __has_builtin(__builtin_amdgcn_global_load_async_to_lds_b128)
#define USE_ASYNC_LDS 1
#else
#define USE_ASYNC_LDS 0
#endif

#if USE_ASYNC_LDS
#if __has_builtin(__builtin_amdgcn_s_wait_asynccnt)
#define ASYNC_WAIT3() __builtin_amdgcn_s_wait_asynccnt(3)
#else
#define ASYNC_WAIT3() asm volatile("s_wait_asynccnt 3" ::: "memory")
#endif
#else
#define ASYNC_WAIT3()
#endif

// ---------------------------------------------------------------------------
// Kernel 0: transpose + convert weights: Wt[w][h][k] = W_w[k][h] as bf16.
// grid = 768 x 256 covers exactly 3*64*1024 elements.
// ---------------------------------------------------------------------------
__global__ __launch_bounds__(256) void prep_weights(
    const float* __restrict__ Wq, const float* __restrict__ Wk,
    const float* __restrict__ Wv, bf16_t* __restrict__ Wt)
{
    int tid = blockIdx.x * 256 + threadIdx.x;     // [0, 3*H*E)
    int w   = tid >> 16;                          // H*E = 65536 per matrix
    int rem = tid & 65535;
    int h   = rem >> 10;
    int k   = rem & 1023;
    const float* src = (w == 0) ? Wq : (w == 1) ? Wk : Wv;
    Wt[tid] = (bf16_t)src[k * H_ + h];
}

// ---------------------------------------------------------------------------
// Stage one 12KB weight slab (192 rows x 32 k-values, bf16) into LDS.
// 768 16-byte chunks; 256 threads x 3 chunks each.
// ---------------------------------------------------------------------------
__device__ __forceinline__ void stage_slab(const bf16_t* __restrict__ Wt,
                                           bf16_t (*sm)[32], int k0, int t)
{
#pragma unroll
    for (int r = 0; r < 3; ++r) {
        const int c    = t + r * 256;     // chunk 0..767
        const int n    = c >> 2;          // weight row 0..191
        const int part = c & 3;           // 16B piece within 64B row-slab
        const bf16_t* g = Wt + (size_t)n * E_ + k0 + part * 8;
        bf16_t* l = &sm[n][part * 8];
#if USE_ASYNC_LDS
        __builtin_amdgcn_global_load_async_to_lds_b128((gv4i*)g, (lv4i*)l, 0, 0);
#else
        *(uint4*)l = *(const uint4*)g;
#endif
    }
}

// Read one 16x16 B-fragment (32 bytes/lane, contiguous) from the LDS slab.
__device__ __forceinline__ v16bf lds_bfrag(const bf16_t (*sm)[32], int n, int kSelB)
{
    struct U { uint4 lo, hi; } u;
    const uint4* p = (const uint4*)&sm[n][kSelB];
    u.lo = p[0];
    u.hi = p[1];
    return __builtin_bit_cast(v16bf, u);
}

// ---------------------------------------------------------------------------
// Kernel 1: fused QKV projection.  Per wave: 16 rows x 192 cols, K-loop E=1024.
// A (idx) converted fp32->bf16 in registers; B slabs double-buffered in LDS
// via async copies (ASYNCcnt) and consumed as ds_load_b128 fragments.
// ---------------------------------------------------------------------------
__global__ __launch_bounds__(256) void qkv_proj(
    const float* __restrict__ X, const bf16_t* __restrict__ Wt,
    const float* __restrict__ bq, const float* __restrict__ bk,
    const float* __restrict__ bv,
    bf16_t* __restrict__ Qb, bf16_t* __restrict__ Kb, bf16_t* __restrict__ Vb)
{
    __shared__ __align__(16) bf16_t smem[2][192][32];   // 24 KB double buffer

    const int t     = threadIdx.x;
    const int lane  = t & 31;
    const int wave  = t >> 5;
    const int m0    = blockIdx.x * 128 + wave * 16;
    const int row   = m0 + (lane & 15);
    const int kSelA = (lane < 16) ? 0 : 16;   // 16-bit A-frag lane K offset
    const int kSelB = (lane < 16) ? 0 : 16;   // 16-bit B-frag lane K offset
    const int cl    = lane & 15;

    v8f acc[12];
#pragma unroll
    for (int i = 0; i < 12; ++i)
#pragma unroll
        for (int j = 0; j < 8; ++j) acc[i][j] = 0.0f;

    const float* arow = X + (size_t)row * E_;

    stage_slab(Wt, smem[0], 0, t);            // prologue copy

    for (int k0 = 0; k0 < E_; k0 += 32) {
        const int cur  = (k0 >> 5) & 1;
        const bool more = (k0 + 32) < E_;
        // Always issue 3 copies so the asynccnt wait is a compile-time imm;
        // the final (unused) slab is drained by s_endpgm's implicit wait.
        stage_slab(Wt, smem[cur ^ 1], more ? (k0 + 32) : 0, t);
        ASYNC_WAIT3();                        // our slab-`cur` copies done
        __syncthreads();                      // everyone's copies visible

        __builtin_prefetch(arow + k0 + 128, 0, 0);   // next A tiles -> caches
        // A fragment: row `row`, K = k0..k0+31, element s at k = k0 + 16*(lane/16) + s (contiguous order)
        v16bf a;
#pragma unroll
        for (int j = 0; j < 8; ++j) {
            a[j]     = (bf16_t)arow[k0 + kSelA + j];
            a[8 + j] = (bf16_t)arow[k0 + kSelA + 8 + j];
        }

        // 12 WMMAs, B-fragments software-pipelined from LDS
        v16bf bcur = lds_bfrag(smem[cur], cl, kSelB);
#pragma unroll
        for (int nt = 0; nt < 12; ++nt) {
            v16bf bnext;
            if (nt < 11) bnext = lds_bfrag(smem[cur], (nt + 1) * 16 + cl, kSelB);
            acc[nt] = __builtin_amdgcn_wmma_f32_16x16x32_bf16(
                false, a, false, bcur, (short)0, acc[nt], false, false);
            bcur = bnext;
        }
        __syncthreads();                      // done reading slab `cur`
    }

    // Store with bias: tiles 0..3 -> Q, 4..7 -> K, 8..11 -> V  (bf16)
    const int rbase = m0 + ((lane < 16) ? 0 : 8);
#pragma unroll
    for (int nt = 0; nt < 12; ++nt) {
        const int gcol = nt * 16 + cl;        // 0..191
        bf16_t* dst; float bias; int col;
        if (gcol < 64)        { dst = Qb; bias = bq[gcol];       col = gcol; }
        else if (gcol < 128)  { dst = Kb; bias = bk[gcol - 64];  col = gcol - 64; }
        else                  { dst = Vb; bias = bv[gcol - 128]; col = gcol - 128; }
#pragma unroll
        for (int i = 0; i < 8; ++i)
            dst[(size_t)(rbase + i) * H_ + col] = (bf16_t)(acc[nt][i] + bias);
    }
}

// ---------------------------------------------------------------------------
// Kernel 2: S[b] = K[b]^T @ V[b]  (64x64 per batch).  64 waves total; each
// wave owns one (batch, 16x16 tile) and reduces over all T -> deterministic.
// ---------------------------------------------------------------------------
__global__ __launch_bounds__(256) void kt_v(
    const bf16_t* __restrict__ Kb, const bf16_t* __restrict__ Vb,
    float* __restrict__ S)
{
    const int lane  = threadIdx.x & 31;
    const int wave  = threadIdx.x >> 5;
    const int gid   = blockIdx.x * 8 + wave;   // 0..63
    const int b     = gid >> 4;
    const int tile  = gid & 15;
    const int h1b   = (tile >> 2) * 16;
    const int h2b   = (tile & 3) * 16;
    const int h1    = h1b + (lane & 15);
    const int h2    = h2b + (lane & 15);
    const int kSelA = (lane < 16) ? 0 : 16;
    const int kSelB = (lane < 16) ? 0 : 16;
    const size_t base = (size_t)b * T_ * H_;

    v8f acc;
#pragma unroll
    for (int j = 0; j < 8; ++j) acc[j] = 0.0f;

    for (int t0 = 0; t0 < T_; t0 += 32) {
        v16bf a, bb;
        // A = K^T : A[h1][t] = K[t][h1]  (strided gathers, tiny & L2-hot)
#pragma unroll
        for (int j = 0; j < 8; ++j) {
            a[j]     = Kb[base + (size_t)(t0 + kSelA + j) * H_ + h1];
            a[8 + j] = Kb[base + (size_t)(t0 + kSelA + 8 + j) * H_ + h1];
        }
#pragma unroll
        for (int j = 0; j < 16; ++j)
            bb[j] = Vb[base + (size_t)(t0 + kSelB + j) * H_ + h2];
        acc = __builtin_amdgcn_wmma_f32_16x16x32_bf16(
            false, a, false, bb, (short)0, acc, false, false);
    }

    const int rb = h1b + ((lane < 16) ? 0 : 8);
#pragma unroll
    for (int i = 0; i < 8; ++i)
        S[b * (H_ * H_) + (rb + i) * H_ + h2] = acc[i];
}

// ---------------------------------------------------------------------------
// Kernel 3: out = scale * Q @ S   ([16384x64] @ [64x64] per batch).
// ---------------------------------------------------------------------------
__global__ __launch_bounds__(256) void q_s(
    const bf16_t* __restrict__ Qb, const float* __restrict__ S,
    float* __restrict__ out)
{
    const int lane  = threadIdx.x & 31;
    const int wave  = threadIdx.x >> 5;
    const int m0    = blockIdx.x * 128 + wave * 16;
    const int b     = (blockIdx.x * 128) / T_;   // whole block in one batch
    const int row   = m0 + (lane & 15);
    const int kSelA = (lane < 16) ? 0 : 16;
    const int kSelB = (lane < 16) ? 0 : 16;
    const float scale = 0.03125f;                // 1024^-0.5

    v8f acc[4];
#pragma unroll
    for (int i = 0; i < 4; ++i)
#pragma unroll
        for (int j = 0; j < 8; ++j) acc[i][j] = 0.0f;

    const bf16_t* qrow = Qb + (size_t)row * H_;
    const float*  Sb   = S + b * (H_ * H_);

    for (int k0 = 0; k0 < H_; k0 += 32) {
        v16bf a;
#pragma unroll
        for (int j = 0; j < 8; ++j) {
            a[j]     = qrow[k0 + kSelA + j];
            a[8 + j] = qrow[k0 + kSelA + 8 + j];
        }
#pragma unroll
        for (int nt = 0; nt < 4; ++nt) {
            const int n = nt * 16 + (lane & 15);
            v16bf bb;
#pragma unroll
            for (int j = 0; j < 16; ++j)
                bb[j] = (bf16_t)Sb[(k0 + kSelB + j) * H_ + n];
            acc[nt] = __builtin_amdgcn_wmma_f32_16x16x32_bf16(
                false, a, false, bb, (short)0, acc[nt], false, false);
        }
    }

    const int rbase = m0 + ((lane < 16) ? 0 : 8);
    const int cl    = lane & 15;
#pragma unroll
    for (int nt = 0; nt < 4; ++nt)
#pragma unroll
        for (int i = 0; i < 8; ++i)
            out[(size_t)(rbase + i) * H_ + nt * 16 + cl] = acc[nt][i] * scale;
}

// ---------------------------------------------------------------------------
extern "C" void kernel_launch(void* const* d_in, const int* in_sizes, int n_in,
                              void* d_out, int out_size, void* d_ws, size_t ws_size,
                              hipStream_t stream) {
    const float* idx = (const float*)d_in[0];
    const float* Wq  = (const float*)d_in[1];
    const float* bq  = (const float*)d_in[2];
    const float* Wk  = (const float*)d_in[3];
    const float* bk  = (const float*)d_in[4];
    const float* Wv  = (const float*)d_in[5];
    const float* bv  = (const float*)d_in[6];
    float* out = (float*)d_out;

    char* ws = (char*)d_ws;
    bf16_t* Wt = (bf16_t*)(ws + WT_OFF);
    bf16_t* Qb = (bf16_t*)(ws + Q_OFF);
    bf16_t* Kb = (bf16_t*)(ws + K_OFF);
    bf16_t* Vb = (bf16_t*)(ws + V_OFF);
    float*  S  = (float*)(ws + S_OFF);

    prep_weights<<<768, 256, 0, stream>>>(Wq, Wk, Wv, Wt);           // 196608 elems
    qkv_proj   <<<M_ / 128, 256, 0, stream>>>(idx, Wt, bq, bk, bv, Qb, Kb, Vb);
    kt_v       <<<8, 256, 0, stream>>>(Kb, Vb, S);                   // 64 waves
    q_s        <<<M_ / 128, 256, 0, stream>>>(Qb, S, out);
}
